// Decoder_22290880266284
// MI455X (gfx1250) — matrix operand order repaired and checked by reading, not compiled, not run on hardware
//
#include <hip/hip_runtime.h>
#include <hip/hip_bf16.h>

// ---------------- problem dimensions (fixed by reference) ----------------
static constexpr int kB  = 4;
static constexpr int kT  = 2048;
static constexpr int kE  = 128;
static constexpr int kH  = 8;
static constexpr int kD  = kH * kE;   // 1024
static constexpr int kF  = 4 * kD;    // 4096
static constexpr int kBT = kB * kT;   // 8192
static constexpr float kScale = 0.08838834764831845f; // 1/sqrt(128)

typedef __attribute__((ext_vector_type(16))) __bf16 v16bf;
typedef __attribute__((ext_vector_type(8)))  float  v8f;
typedef __attribute__((ext_vector_type(4)))  __bf16 bf16x4;

// ---------------- WMMA helpers (CDNA5 16x16x32 bf16, f32 accum) ----------
__device__ __forceinline__ v8f wmma_bf16(v16bf a, v16bf b, v8f c) {
  // (neg_a, A, neg_b, B, c_mod, C, reuse_a, reuse_b)
  return __builtin_amdgcn_wmma_f32_16x16x32_bf16(false, a, false, b, (short)0, c,
                                                 false, false);
}

// A fragment: 16x32 bf16 from row-major src (rows=M, cols=K).
// lanes 0-15: M=lane, K in {0..15}; lanes 16-31: K in {16..31}
__device__ __forceinline__ v16bf load_a_frag(const __bf16* base, int ld, int rbase,
                                             int kbase, int lane) {
  const int hi = (lane >> 4) & 1;
  const int m  = lane & 15;
  const __bf16* p = base + (long)(rbase + m) * ld + kbase;
  v16bf a;
#pragma unroll
  for (int v = 0; v < 8; ++v) {
    const int k0 = hi * 16 + 2 * v;
    a[2 * v]     = p[k0];
    a[2 * v + 1] = p[k0 + 1];
  }
  return a;
}

// B fragment: 32x16 bf16 from row-major src (rows=K, cols=N).
// lanes 0-15 cover K=0..15, lanes 16-31 cover K=16..31; N = lane&15
__device__ __forceinline__ v16bf load_b_frag(const __bf16* base, int ld, int kbase,
                                             int nbase, int lane) {
  const int hi = (lane >> 4) & 1;
  const int n  = lane & 15;
  const __bf16* p = base + nbase + n + (long)(kbase + hi * 16) * ld;
  v16bf b;
#pragma unroll
  for (int v = 0; v < 8; ++v) {
    b[2 * v]     = p[(2 * v) * ld];
    b[2 * v + 1] = p[(2 * v + 1) * ld];
  }
  return b;
}

// B fragment where the logical B is the transpose of the stored tile:
// B[k][n] = src[nbase+n][kbase+k]  (contiguous 16 bf16 per lane)
__device__ __forceinline__ v16bf load_bT_frag(const __bf16* base, int ld, int kbase,
                                              int nbase, int lane) {
  const int hi = (lane >> 4) & 1;
  const int n  = lane & 15;
  const __bf16* p = base + (long)(nbase + n) * ld + kbase + hi * 16;
  v16bf b;
#pragma unroll
  for (int v = 0; v < 8; ++v) {
    b[2 * v]     = p[2 * v];
    b[2 * v + 1] = p[2 * v + 1];
  }
  return b;
}

// ---------------- LayerNorm (fp32 in -> bf16 out) -----------------------
__global__ void ln_bf16_kernel(const float* __restrict__ x, const float* __restrict__ g,
                               const float* __restrict__ bb, __bf16* __restrict__ out,
                               int C) {
  const long row = blockIdx.x;
  const float* xr = x + row * C;
  __bf16* orow = out + row * C;
  float s = 0.f, s2 = 0.f;
  for (int i = threadIdx.x; i < C; i += blockDim.x) {
    float v = xr[i]; s += v; s2 += v * v;
  }
#pragma unroll
  for (int o = 16; o > 0; o >>= 1) {
    s  += __shfl_xor(s, o, 32);
    s2 += __shfl_xor(s2, o, 32);
  }
  __shared__ float sh[2][32];
  const int wid = threadIdx.x >> 5, ln = threadIdx.x & 31;
  const int nw = blockDim.x >> 5;
  if (ln == 0) { sh[0][wid] = s; sh[1][wid] = s2; }
  __syncthreads();
  s = 0.f; s2 = 0.f;
  for (int i = 0; i < nw; ++i) { s += sh[0][i]; s2 += sh[1][i]; }
  const float mu   = s / C;
  const float var  = s2 / C - mu * mu;
  const float rstd = rsqrtf(var + 1e-5f);
  for (int i = threadIdx.x; i < C; i += blockDim.x)
    orow[i] = (__bf16)((xr[i] - mu) * rstd * g[i] + bb[i]);
}

// ---------------- fp32 -> bf16 weight conversion ------------------------
__global__ void cvt_f32_bf16_kernel(const float* __restrict__ in,
                                    __bf16* __restrict__ out, long n) {
  long i = (long)blockIdx.x * blockDim.x + threadIdx.x;
  if (i < n) out[i] = (__bf16)in[i];
}

// ---------------- generic tiled GEMM: C = act(A*B + bias) ----------------
// Block tile 128x128, 8 waves, each wave 16 rows x 128 cols (8 wmma tiles).
// K chunked by 64 through LDS. Optional batching: z -> A[(z/div)*sA],
// B[(z%div)*sB], bias[(z%div)*sBias], C[z*sC].
template <int ACT, int OUTF32>
__global__ void __launch_bounds__(256) gemm_bf16_kernel(
    const __bf16* __restrict__ A, const __bf16* __restrict__ Bw,
    const float* __restrict__ bias, void* __restrict__ Cout,
    int K, int lda, int ldb, int ldc,
    long sA, long sB, long sBias, long sC, int divh) {
  const int z = blockIdx.z;
  A    += (long)(z / divh) * sA;
  Bw   += (long)(z % divh) * sB;
  bias += (long)(z % divh) * sBias;
  float*  Cf = (float*)Cout  + (long)z * sC;
  __bf16* Cb = (__bf16*)Cout + (long)z * sC;

  const int row0 = blockIdx.x * 128;
  const int col0 = blockIdx.y * 128;

  __shared__ __bf16 As[128][72];    // 128x64 + pad
  __shared__ __bf16 Bs[64][136];    // 64x128 + pad

  const int tid = threadIdx.x;
  const int w = tid >> 5, lane = tid & 31;

  v8f acc[8];
  v8f zero = {};
#pragma unroll
  for (int j = 0; j < 8; ++j) acc[j] = zero;

  for (int kk = 0; kk < K; kk += 64) {
    __syncthreads();
    for (int i = tid; i < 128 * 16; i += 256) {        // A: 128 rows x 16 vec4
      int r = i >> 4, c = (i & 15) * 4;
      *(bf16x4*)&As[r][c] = *(const bf16x4*)(A + (long)(row0 + r) * lda + kk + c);
    }
    for (int i = tid; i < 64 * 32; i += 256) {         // B: 64 rows x 32 vec4
      int r = i >> 5, c = (i & 31) * 4;
      *(bf16x4*)&Bs[r][c] = *(const bf16x4*)(Bw + (long)(kk + r) * ldb + col0 + c);
    }
    __syncthreads();
#pragma unroll
    for (int ks = 0; ks < 64; ks += 32) {
      v16bf a = load_a_frag(&As[0][0], 72, w * 16, ks, lane);
#pragma unroll
      for (int j = 0; j < 8; ++j) {
        v16bf bfr = load_b_frag(&Bs[0][0], 136, ks, j * 16, lane);
        acc[j] = wmma_bf16(a, bfr, acc[j]);
      }
    }
  }

  const int hi = lane >> 4, n = lane & 15;
#pragma unroll
  for (int j = 0; j < 8; ++j) {
    const int col = col0 + j * 16 + n;
    const float bv = bias[col];
#pragma unroll
    for (int r = 0; r < 8; ++r) {
      const int row = row0 + w * 16 + r + hi * 8;
      float val = acc[j][r] + bv;
      if (ACT == 1) val = 0.5f * val * (1.0f + erff(val * 0.70710678118654752f));
      if (OUTF32) Cf[(long)row * ldc + col] = val;
      else        Cb[(long)row * ldc + col] = (__bf16)val;
    }
  }
}

// ---------------- attention pass 1: per-KEY-column softmax stats ---------
// softmax is over the QUERY axis: for column s, over t in [s, T).
// Each block handles 128 columns of one (b,h); wave w owns 16 columns.
__global__ void __launch_bounds__(256) attn_stats_kernel(
    const __bf16* __restrict__ q, const __bf16* __restrict__ k,
    float* __restrict__ colmax, float* __restrict__ colsum) {
  const int s0 = blockIdx.x * 128;
  const int h = blockIdx.y, b = blockIdx.z;
  const long bh = (long)b * kH + h;
  const __bf16* qbh = q + bh * (long)kT * kE;
  const __bf16* kbh = k + bh * (long)kT * kE;

  __shared__ __bf16 k_lds[128][kE + 8];
  __shared__ __bf16 q_lds[16][kE + 8];

  const int tid = threadIdx.x, w = tid >> 5, lane = tid & 31;
  const int hi = lane >> 4, n = lane & 15;
  const int swave = s0 + w * 16;

  for (int i = tid; i < 128 * 32; i += 256) {
    int r = i >> 5, c = (i & 31) * 4;
    *(bf16x4*)&k_lds[r][c] = *(const bf16x4*)(kbh + (long)(s0 + r) * kE + c);
  }

  float mcol = -1e30f, zcol = 0.f;

  for (int t = s0; t < kT; t += 16) {
    __syncthreads();
    for (int i = tid; i < 16 * 32; i += 256) {
      int r = i >> 5, c = (i & 31) * 4;
      *(bf16x4*)&q_lds[r][c] = *(const bf16x4*)(qbh + (long)(t + r) * kE + c);
    }
    __syncthreads();

    v8f sc = {};
#pragma unroll
    for (int kk = 0; kk < kE; kk += 32) {
      v16bf a  = load_a_frag(&q_lds[0][0], kE + 8, 0, kk, lane);
      v16bf bb = load_bT_frag(&k_lds[0][0], kE + 8, kk, w * 16, lane);  // k^T
      sc = wmma_bf16(a, bb, sc);
    }

    const int scol = swave + n;
    float tm = -1e30f;
    float pv[8];
#pragma unroll
    for (int r = 0; r < 8; ++r) {
      const int trow = t + r + hi * 8;
      const float v = sc[r] * kScale;
      const bool ok = (scol <= trow);           // causal: valid when t >= s
      pv[r] = ok ? v : -1e30f;
      if (ok) tm = fmaxf(tm, v);
    }
    tm = fmaxf(tm, __shfl_xor(tm, 16, 32));     // merge the two lane-halves
    const float newm = fmaxf(mcol, tm);
    float zadd = 0.f;
#pragma unroll
    for (int r = 0; r < 8; ++r) {
      const int trow = t + r + hi * 8;
      if (scol <= trow) zadd += expf(pv[r] - newm);
    }
    zadd += __shfl_xor(zadd, 16, 32);
    zcol = zcol * expf(mcol - newm) + zadd;
    mcol = newm;
  }

  if (hi == 0) {
    const long idx = bh * kT + swave + n;
    colmax[idx] = mcol;
    colsum[idx] = zcol;
  }
}

// ---------------- attention pass 2: out = P * V, P through LDS -----------
// Block handles 128 query rows of one (b,h); wave w owns 16 rows x 128 f.
__global__ void __launch_bounds__(256) attn_out_kernel(
    const __bf16* __restrict__ q, const __bf16* __restrict__ k,
    const __bf16* __restrict__ v, const float* __restrict__ colmax,
    const float* __restrict__ colsum, float* __restrict__ concat) {
  const int t0 = blockIdx.x * 128;
  const int h = blockIdx.y, b = blockIdx.z;
  const long bh = (long)b * kH + h;
  const __bf16* qbh = q + bh * (long)kT * kE;
  const __bf16* kbh = k + bh * (long)kT * kE;
  const __bf16* vbh = v + bh * (long)kT * kE;

  __shared__ __bf16 q_lds[128][kE + 8];
  __shared__ __bf16 kv_lds[2][32][kE + 8];   // [0]=K rows, [1]=V rows
  __shared__ __bf16 p_lds[8][16][40];        // per-wave P tile 16x32 (+pad)

  const int tid = threadIdx.x, w = tid >> 5, lane = tid & 31;
  const int hi = lane >> 4, n = lane & 15;

  for (int i = tid; i < 128 * 32; i += 256) {
    int r = i >> 5, c = (i & 31) * 4;
    *(bf16x4*)&q_lds[r][c] = *(const bf16x4*)(qbh + (long)(t0 + r) * kE + c);
  }

  v8f acc[8];
  v8f zero = {};
#pragma unroll
  for (int j = 0; j < 8; ++j) acc[j] = zero;

  const int send = t0 + 128;                  // causal upper bound for block
  for (int s = 0; s < send; s += 32) {
    __syncthreads();
    for (int i = tid; i < 2 * 32 * 32; i += 256) {
      int which = i >> 10, rem = i & 1023;
      int r = rem >> 5, c = (rem & 31) * 4;
      const __bf16* src = which ? vbh : kbh;
      *(bf16x4*)&kv_lds[which][r][c] = *(const bf16x4*)(src + (long)(s + r) * kE + c);
    }
    __syncthreads();

#pragma unroll
    for (int sub = 0; sub < 2; ++sub) {
      v8f sc = {};
#pragma unroll
      for (int kk = 0; kk < kE; kk += 32) {
        v16bf a  = load_a_frag(&q_lds[0][0], kE + 8, w * 16, kk, lane);
        v16bf bb = load_bT_frag(&kv_lds[0][0][0], kE + 8, kk, sub * 16, lane);
        sc = wmma_bf16(a, bb, sc);
      }
      const int scol = s + sub * 16 + n;
      const float ms = colmax[bh * kT + scol];
      const float zs = colsum[bh * kT + scol];
      const float inv = (zs > 0.f) ? 1.f / zs : 0.f;
#pragma unroll
      for (int r = 0; r < 8; ++r) {
        const int trow = t0 + w * 16 + r + hi * 8;
        float p = 0.f;
        if (scol <= trow) p = expf(sc[r] * kScale - ms) * inv;
        p_lds[w][r + hi * 8][sub * 16 + n] = (__bf16)p;
      }
    }

    // P(16x32) * V(32x128) -> accumulate (wave-private LDS, in-order DS ops)
    v16bf a = load_a_frag(&p_lds[w][0][0], 40, 0, 0, lane);
#pragma unroll
    for (int j = 0; j < 8; ++j) {
      v16bf bb = load_b_frag(&kv_lds[1][0][0], kE + 8, 0, j * 16, lane);
      acc[j] = wmma_bf16(a, bb, acc[j]);
    }
  }

#pragma unroll
  for (int j = 0; j < 8; ++j) {
#pragma unroll
    for (int r = 0; r < 8; ++r) {
      const int trow = t0 + w * 16 + r + hi * 8;
      const int col = h * kE + j * 16 + n;                 // head concat
      concat[((long)b * kT + trow) * kD + col] = acc[j][r];
    }
  }
}

// ---------------- host-side orchestration --------------------------------
extern "C" void kernel_launch(void* const* d_in, const int* in_sizes, int n_in,
                              void* d_out, int out_size, void* d_ws, size_t ws_size,
                              hipStream_t stream) {
  (void)in_sizes; (void)n_in; (void)out_size; (void)ws_size;
  const float* X     = (const float*)d_in[0];
  const float* ln1_g = (const float*)d_in[1];
  const float* ln1_b = (const float*)d_in[2];
  const float* Wq    = (const float*)d_in[3];
  const float* bq    = (const float*)d_in[4];
  const float* Wk    = (const float*)d_in[5];
  const float* bk    = (const float*)d_in[6];
  const float* Wv    = (const float*)d_in[7];
  const float* bv    = (const float*)d_in[8];
  const float* ln2_g = (const float*)d_in[9];
  const float* ln2_b = (const float*)d_in[10];
  const float* W1    = (const float*)d_in[11];
  const float* b1    = (const float*)d_in[12];
  const float* W2    = (const float*)d_in[13];
  const float* b2    = (const float*)d_in[14];

  char* ws = (char*)d_ws;
  size_t off = 0;
  auto take = [&](size_t bytes) -> void* {
    void* p = ws + off;
    off = (off + bytes + 255) & ~(size_t)255;
    return p;
  };
  // total workspace ~175 MB
  __bf16* Xn   = (__bf16*)take((size_t)kBT * kE * 2);
  __bf16* Wqb  = (__bf16*)take((size_t)kH * kE * kE * 2);
  __bf16* Wkb  = (__bf16*)take((size_t)kH * kE * kE * 2);
  __bf16* Wvb  = (__bf16*)take((size_t)kH * kE * kE * 2);
  __bf16* W1b  = (__bf16*)take((size_t)kD * kF * 2);
  __bf16* W2b  = (__bf16*)take((size_t)kF * kE * 2);
  __bf16* qb   = (__bf16*)take((size_t)kB * kH * kT * kE * 2);
  __bf16* kb   = (__bf16*)take((size_t)kB * kH * kT * kE * 2);
  __bf16* vb   = (__bf16*)take((size_t)kB * kH * kT * kE * 2);
  float*  cmax = (float*)take((size_t)kB * kH * kT * 4);
  float*  csum = (float*)take((size_t)kB * kH * kT * 4);
  float*  cat  = (float*)take((size_t)kBT * kD * 4);
  __bf16* h2   = (__bf16*)take((size_t)kBT * kD * 2);
  __bf16* mid  = (__bf16*)take((size_t)kBT * kF * 2);

  // 1) LN1 over E=128 -> bf16
  ln_bf16_kernel<<<kBT, 128, 0, stream>>>(X, ln1_g, ln1_b, Xn, kE);

  // 2) weights -> bf16
  {
    long n = (long)kH * kE * kE;
    dim3 g((unsigned)((n + 255) / 256));
    cvt_f32_bf16_kernel<<<g, 256, 0, stream>>>(Wq, Wqb, n);
    cvt_f32_bf16_kernel<<<g, 256, 0, stream>>>(Wk, Wkb, n);
    cvt_f32_bf16_kernel<<<g, 256, 0, stream>>>(Wv, Wvb, n);
  }
  {
    long n = (long)kD * kF;
    cvt_f32_bf16_kernel<<<dim3((unsigned)((n + 255) / 256)), 256, 0, stream>>>(W1, W1b, n);
  }
  {
    long n = (long)kF * kE;
    cvt_f32_bf16_kernel<<<dim3((unsigned)((n + 255) / 256)), 256, 0, stream>>>(W2, W2b, n);
  }

  // 3) QKV projections: batched over z = b*H + h (A indexed by b, B/bias by h)
  dim3 gqkv(kT / 128, kE / 128, kB * kH);
  gemm_bf16_kernel<0, 0><<<gqkv, 256, 0, stream>>>(
      Xn, Wqb, bq, qb, kE, kE, kE, kE,
      (long)kT * kE, (long)kE * kE, (long)kE, (long)kT * kE, kH);
  gemm_bf16_kernel<0, 0><<<gqkv, 256, 0, stream>>>(
      Xn, Wkb, bk, kb, kE, kE, kE, kE,
      (long)kT * kE, (long)kE * kE, (long)kE, (long)kT * kE, kH);
  gemm_bf16_kernel<0, 0><<<gqkv, 256, 0, stream>>>(
      Xn, Wvb, bv, vb, kE, kE, kE, kE,
      (long)kT * kE, (long)kE * kE, (long)kE, (long)kT * kE, kH);

  // 4) column-softmax stats, 5) attention output into concat layout
  dim3 gat(kT / 128, kH, kB);
  attn_stats_kernel<<<gat, 256, 0, stream>>>(qb, kb, cmax, csum);
  attn_out_kernel<<<gat, 256, 0, stream>>>(qb, kb, vb, cmax, csum, cat);

  // 6) LN2 over D=1024 -> bf16
  ln_bf16_kernel<<<kBT, 256, 0, stream>>>(cat, ln2_g, ln2_b, h2, kD);

  // 7) MLP1 with exact GELU -> bf16
  gemm_bf16_kernel<1, 0><<<dim3(kBT / 128, kF / 128, 1), 256, 0, stream>>>(
      h2, W1b, b1, mid, kD, kD, kF, kF, 0, 0, 0, 0, 1);

  // 8) MLP2 -> fp32 output
  gemm_bf16_kernel<0, 1><<<dim3(kBT / 128, kE / 128, 1), 256, 0, stream>>>(
      mid, W2b, b2, d_out, kF, kF, kE, kE, 0, 0, 0, 0, 1);
}
